// DotProductAttention_14044543058267
// MI455X (gfx1250) — matrix operand order repaired and checked by reading, not compiled, not run on hardware
//
#include <hip/hip_runtime.h>
#include <hip/hip_bf16.h>
#include <math.h>

typedef __bf16 bf16;
typedef __attribute__((ext_vector_type(8)))  bf16  v8bf;
typedef __attribute__((ext_vector_type(16))) bf16  v16bf;
typedef __attribute__((ext_vector_type(8)))  float v8f;

#define DIM      1024
#define NHEAD    16
#define HD       64
#define SEQ      1025
#define BATCH    8
#define M_ROWS   (BATCH * SEQ)   // 8200
#define M_PAD    8448            // 33 * 256 (GEMM row blocking)
#define SQ_PAD   1040            // padded q rows (65*16)
#define SK_PAD   1056            // padded kv rows (33*32)
#define NEG_INF  (-1.0e30f)

// ---------------- WMMA fragment helpers (wave32, 16x16x32 bf16) ----------------
// A (16xK=32, 16-bit): lane<16 -> row=lane, elems {k0..k0+7, k0+16..k0+23}
//                      lane>=16 -> row=lane-16, elems {k0+8..+15, k0+24..+31}
__device__ __forceinline__ v16bf load_a_frag(const bf16* A, int lda, int m0, int k0, int lane) {
    int hi = lane >> 4;
    int m  = m0 + (lane & 15);
    const bf16* p = A + (size_t)m * lda + k0 + hi * 8;
    v16bf a;
    ((v8bf*)&a)[0] = *(const v8bf*)(p);
    ((v8bf*)&a)[1] = *(const v8bf*)(p + 16);
    return a;
}
// B (K=32 x 16, 16-bit): lane = n + 16*khalf holds B[k][n] for
// k in {k0+khalf*8..+7, k0+16+khalf*8..+7} (W row n0+n; W is [N,K] row-major).
__device__ __forceinline__ v16bf load_b_frag(const bf16* W, int ldw, int n0, int k0, int lane) {
    int khalf = lane >> 4;
    int n = lane & 15;
    const bf16* p = W + (size_t)(n0 + n) * ldw + k0 + khalf * 8;
    v16bf b;
    ((v8bf*)&b)[0] = *(const v8bf*)(p);
    ((v8bf*)&b)[1] = *(const v8bf*)(p + 16);
    return b;
}

__device__ __forceinline__ v8f zero_v8f() {
    v8f z;
#pragma unroll
    for (int i = 0; i < 8; ++i) z[i] = 0.0f;
    return z;
}

#define WMMA_BF16(a, b, c) \
    __builtin_amdgcn_wmma_f32_16x16x32_bf16(false, (a), false, (b), (short)0, (c), false, false)

// 32x64 output tile per wave: 2 A-frags x 4 B-frags = 8 WMMAs per K-step.
__device__ __forceinline__ void gemm_32x64(const bf16* __restrict__ A, const bf16* __restrict__ W,
                                           int kdim, int m0, int n0, int lane, v8f c[2][4]) {
    for (int k0 = 0; k0 < kdim; k0 += 32) {
        if (k0 + 64 < kdim) {
            // speculative next-next-chunk prefetch (global_prefetch_b8): one line per lane
            __builtin_prefetch(A + (size_t)(m0 + lane) * kdim + k0 + 64, 0, 1);
            __builtin_prefetch(W + (size_t)(n0 + lane) * kdim + k0 + 64, 0, 1);
            __builtin_prefetch(W + (size_t)(n0 + 32 + lane) * kdim + k0 + 64, 0, 1);
        }
        v16bf a0 = load_a_frag(A, kdim, m0, k0, lane);
        v16bf a1 = load_a_frag(A, kdim, m0 + 16, k0, lane);
#pragma unroll
        for (int t = 0; t < 4; ++t) {
            v16bf b = load_b_frag(W, kdim, n0 + 16 * t, k0, lane);
            c[0][t] = WMMA_BF16(a0, b, c[0][t]);
            c[1][t] = WMMA_BF16(a1, b, c[1][t]);
        }
    }
}

// ---------------- Kernel 0: f32 -> bf16 convert ----------------
__global__ void cvt_f32_bf16(const float* __restrict__ in, bf16* __restrict__ out, int n) {
    int i = blockIdx.x * blockDim.x + threadIdx.x;
    if (i < n) out[i] = (bf16)in[i];
}

// ---------------- Kernel 1: QKV GEMM + scatter epilogue ----------------
// Y[m, 0..3071] = X @ Wqkv^T + b ; scatter into q [b,h,1040,64] (pre-scaled 1/8),
// k [b,h,1056,64], v^T [b,h,64,1056]
__global__ void qkv_gemm_kernel(const bf16* __restrict__ X, const bf16* __restrict__ W,
                                const float* __restrict__ bias,
                                bf16* __restrict__ qb, bf16* __restrict__ kb,
                                bf16* __restrict__ vtb) {
    const int lane = threadIdx.x & 31;
    const int wave = threadIdx.x >> 5;
    const int m0 = blockIdx.x * 256 + wave * 32;
    const int n0 = blockIdx.y * 64;
    v8f c[2][4];
#pragma unroll
    for (int u = 0; u < 2; ++u)
#pragma unroll
        for (int t = 0; t < 4; ++t) c[u][t] = zero_v8f();
    gemm_32x64(X, W, DIM, m0, n0, lane, c);

    const int hi = lane >> 4, col = lane & 15;
#pragma unroll
    for (int u = 0; u < 2; ++u) {
#pragma unroll
        for (int t = 0; t < 4; ++t) {
            int n = n0 + t * 16 + col;
            float bv = bias[n];
            int which = n >> 10;
            int hc = n & 1023;
            int hh = hc >> 6, cd = hc & 63;
#pragma unroll
            for (int r = 0; r < 8; ++r) {
                int m = m0 + u * 16 + r + 8 * hi;
                if (m < M_ROWS) {
                    float val = c[u][t][r] + bv;
                    int bb = m / SEQ, s = m - bb * SEQ;
                    size_t bh = (size_t)(bb * NHEAD + hh);
                    if (which == 0)      qb[(bh * SQ_PAD + s) * HD + cd] = (bf16)(val * 0.125f);
                    else if (which == 1) kb[(bh * SK_PAD + s) * HD + cd] = (bf16)val;
                    else                 vtb[(bh * HD + cd) * SK_PAD + s] = (bf16)val;
                }
            }
        }
    }
}

// ---------------- Kernel 2: relative-position bias gather ----------------
// bias_buf[h][i][j] = table[idx[i*1025+j]*16 + h]  (bf16, cols padded to 1056)
__global__ void bias_gather_kernel(const int* __restrict__ idx, const float* __restrict__ table,
                                   bf16* __restrict__ out) {
    int i = blockIdx.x;  // 0..1024
    int h = blockIdx.y;  // 0..15
    for (int j = threadIdx.x; j < SEQ; j += blockDim.x) {
        int id = idx[i * SEQ + j];
        out[((size_t)h * SEQ + i) * SK_PAD + j] = (bf16)table[id * NHEAD + h];
    }
}

// ---------------- Kernel 3: flash attention, one wave per (b,h,16-row q tile) ----------------
__global__ void attn_kernel(const bf16* __restrict__ q, const bf16* __restrict__ k,
                            const bf16* __restrict__ vt, const bf16* __restrict__ bias,
                            bf16* __restrict__ aout) {
    __shared__ bf16 lds_p[16 * 32];  // probs C-frag -> A-frag staging
    const int lane = threadIdx.x;
    const int qt = blockIdx.x, h = blockIdx.y, bidx = blockIdx.z;
    const int q0 = qt * 16;
    const int hi = lane >> 4, lcol = lane & 15;

    const bf16* qp = q  + (size_t)(bidx * NHEAD + h) * SQ_PAD * HD;
    const bf16* kp = k  + (size_t)(bidx * NHEAD + h) * SK_PAD * HD;
    const bf16* vp = vt + (size_t)(bidx * NHEAD + h) * HD * SK_PAD;
    const bf16* bp = bias + (size_t)h * SEQ * SK_PAD;

    // preload q A-fragments for both c-chunks (hd=64 -> K=32 x2); q pre-scaled by 1/sqrt(hd)
    v16bf qa0 = load_a_frag(qp, HD, q0, 0, lane);
    v16bf qa1 = load_a_frag(qp, HD, q0, 32, lane);

    v8f o[4] = {zero_v8f(), zero_v8f(), zero_v8f(), zero_v8f()};
    float mrow[8], lrow[8];
#pragma unroll
    for (int r = 0; r < 8; ++r) { mrow[r] = NEG_INF; lrow[r] = 0.0f; }

    for (int cc = 0; cc < 33; ++cc) {
        const int kbase = cc * 32;
        if (cc < 32) {
            // prefetch next chunk: 32 K rows (one line each) + next V columns
            __builtin_prefetch(kp + (size_t)(kbase + 32 + lane) * HD, 0, 1);
            __builtin_prefetch(vp + (size_t)lane * SK_PAD + kbase + 32, 0, 1);
            __builtin_prefetch(vp + (size_t)(32 + lane) * SK_PAD + kbase + 32, 0, 1);
        }
        // scores tile 16x32: two 16x16 C-frags
        v8f s0 = zero_v8f(), s1 = zero_v8f();
        s0 = WMMA_BF16(qa0, load_b_frag(kp, HD, kbase, 0, lane), s0);
        s0 = WMMA_BF16(qa1, load_b_frag(kp, HD, kbase, 32, lane), s0);
        s1 = WMMA_BF16(qa0, load_b_frag(kp, HD, kbase + 16, 0, lane), s1);
        s1 = WMMA_BF16(qa1, load_b_frag(kp, HD, kbase + 16, 32, lane), s1);

        const int colA = kbase + lcol;
        const int colB = colA + 16;
        const int rbase = q0 + 8 * hi;
#pragma unroll
        for (int r = 0; r < 8; ++r) {
            int rowc = rbase + r;
            if (rowc > SEQ - 1) rowc = SEQ - 1;
            float t0 = s0[r] + (float)bp[(size_t)rowc * SK_PAD + colA];
            float t1 = s1[r] + (float)bp[(size_t)rowc * SK_PAD + colB];
            if (colA >= SEQ) t0 = NEG_INF;
            if (colB >= SEQ) t1 = NEG_INF;
            // row max across the 16 lanes holding this row (xor<16 stays in half)
            float mx = fmaxf(t0, t1);
            mx = fmaxf(mx, __shfl_xor(mx, 1));
            mx = fmaxf(mx, __shfl_xor(mx, 2));
            mx = fmaxf(mx, __shfl_xor(mx, 4));
            mx = fmaxf(mx, __shfl_xor(mx, 8));
            float mnew = fmaxf(mrow[r], mx);
            float scale = __expf(mrow[r] - mnew);
            float p0 = __expf(t0 - mnew);
            float p1 = __expf(t1 - mnew);
            float ps = p0 + p1;
            ps += __shfl_xor(ps, 1);
            ps += __shfl_xor(ps, 2);
            ps += __shfl_xor(ps, 4);
            ps += __shfl_xor(ps, 8);
            lrow[r] = lrow[r] * scale + ps;
            mrow[r] = mnew;
#pragma unroll
            for (int t = 0; t < 4; ++t) o[t][r] *= scale;
            lds_p[(r + 8 * hi) * 32 + lcol]      = (bf16)p0;
            lds_p[(r + 8 * hi) * 32 + lcol + 16] = (bf16)p1;
        }
        // re-read probs in A-fragment layout (single wave: DS in-order, no barrier)
        const bf16* pp = &lds_p[lcol * 32 + hi * 8];
        v16bf pa;
        ((v8bf*)&pa)[0] = *(const v8bf*)(pp);
        ((v8bf*)&pa)[1] = *(const v8bf*)(pp + 16);
        // O += P @ V : B-frags contiguous from transposed V
#pragma unroll
        for (int t = 0; t < 4; ++t) {
            v16bf vb = load_b_frag(vp, SK_PAD, t * 16, kbase, lane);
            o[t] = WMMA_BF16(pa, vb, o[t]);
        }
    }

    // finalize: O/l, write bf16 into [b, s, h*64+c] layout
#pragma unroll
    for (int r = 0; r < 8; ++r) {
        int row = q0 + r + 8 * hi;
        if (row < SEQ) {
            float inv = 1.0f / lrow[r];
            size_t base = ((size_t)(bidx * SEQ + row)) * DIM + h * HD;
#pragma unroll
            for (int t = 0; t < 4; ++t)
                aout[base + t * 16 + lcol] = (bf16)(o[t][r] * inv);
        }
    }
}

// ---------------- Kernel 4: output projection -> f32 d_out ----------------
__global__ void proj_gemm_kernel(const bf16* __restrict__ A, const bf16* __restrict__ W,
                                 const float* __restrict__ bias, float* __restrict__ out) {
    const int lane = threadIdx.x & 31;
    const int wave = threadIdx.x >> 5;
    const int m0 = blockIdx.x * 256 + wave * 32;
    const int n0 = blockIdx.y * 64;
    v8f c[2][4];
#pragma unroll
    for (int u = 0; u < 2; ++u)
#pragma unroll
        for (int t = 0; t < 4; ++t) c[u][t] = zero_v8f();
    gemm_32x64(A, W, DIM, m0, n0, lane, c);

    const int hi = lane >> 4, col = lane & 15;
#pragma unroll
    for (int u = 0; u < 2; ++u) {
#pragma unroll
        for (int t = 0; t < 4; ++t) {
            int n = n0 + t * 16 + col;
            float bv = bias[n];
#pragma unroll
            for (int r = 0; r < 8; ++r) {
                int m = m0 + u * 16 + r + 8 * hi;
                if (m < M_ROWS) out[(size_t)m * DIM + n] = c[u][t][r] + bv;
            }
        }
    }
}

// ---------------- Host launch ----------------
extern "C" void kernel_launch(void* const* d_in, const int* in_sizes, int n_in,
                              void* d_out, int out_size, void* d_ws, size_t ws_size,
                              hipStream_t stream) {
    (void)in_sizes; (void)n_in; (void)out_size; (void)ws_size;
    const float* x      = (const float*)d_in[0];
    const float* qkv_w  = (const float*)d_in[1];
    const float* qkv_b  = (const float*)d_in[2];
    const float* proj_w = (const float*)d_in[3];
    const float* proj_b = (const float*)d_in[4];
    const float* table  = (const float*)d_in[5];
    const int*   relidx = (const int*)d_in[6];
    float* out = (float*)d_out;

    char* ws = (char*)d_ws;
    size_t off = 0;
    bf16* Xb   = (bf16*)(ws + off); off += (size_t)M_PAD * DIM * 2;                 // 17.3 MB
    bf16* Wq   = (bf16*)(ws + off); off += (size_t)3 * DIM * DIM * 2;               //  6.3 MB
    bf16* Wp   = (bf16*)(ws + off); off += (size_t)DIM * DIM * 2;                   //  2.1 MB
    bf16* Qb   = (bf16*)(ws + off); off += (size_t)BATCH * NHEAD * SQ_PAD * HD * 2; // 17.0 MB
    bf16* Kb   = (bf16*)(ws + off); off += (size_t)BATCH * NHEAD * SK_PAD * HD * 2; // 17.3 MB
    bf16* Vtb  = (bf16*)(ws + off); off += (size_t)BATCH * NHEAD * HD * SK_PAD * 2; // 17.3 MB
    bf16* Bias = (bf16*)(ws + off); off += (size_t)NHEAD * SEQ * SK_PAD * 2;        // 34.6 MB
    bf16* Ab   = (bf16*)(ws + off); off += (size_t)M_PAD * DIM * 2;                 // 17.3 MB

    // stage bf16 operands
    {
        int n = M_ROWS * DIM;
        cvt_f32_bf16<<<(n + 255) / 256, 256, 0, stream>>>(x, Xb, n);
        n = 3 * DIM * DIM;
        cvt_f32_bf16<<<(n + 255) / 256, 256, 0, stream>>>(qkv_w, Wq, n);
        n = DIM * DIM;
        cvt_f32_bf16<<<(n + 255) / 256, 256, 0, stream>>>(proj_w, Wp, n);
    }
    // rel-pos bias gather (shared across batch)
    bias_gather_kernel<<<dim3(SEQ, NHEAD), 128, 0, stream>>>(relidx, table, Bias);
    // fused QKV GEMM: 256 rows x 64 cols per block
    qkv_gemm_kernel<<<dim3(M_PAD / 256, (3 * DIM) / 64), 256, 0, stream>>>(Xb, Wq, qkv_b,
                                                                           Qb, Kb, Vtb);
    // flash attention
    attn_kernel<<<dim3(SQ_PAD / 16, NHEAD, BATCH), 32, 0, stream>>>(Qb, Kb, Vtb, Bias, Ab);
    // output projection
    proj_gemm_kernel<<<dim3(M_PAD / 256, DIM / 64), 256, 0, stream>>>(Ab, Wp, proj_b, out);
}
